// TemporalGraphNetwork_85444079387010
// MI455X (gfx1250) — matrix operand order repaired and checked by reading, not compiled, not run on hardware
//
#include <hip/hip_runtime.h>
#include <hip/hip_bf16.h>
#include <math.h>

typedef __attribute__((ext_vector_type(16))) _Float16 v16h;
typedef __attribute__((ext_vector_type(8)))  float    v8f;

#define BB    8
#define SS    512
#define NNODE 1024
#define NEDGE 2048
#define MAXN  4096
#define MAXE  8192
#define MEMD  64
#define TENCD 32
#define EEMBD 128
#define OUTD  64
#define MSGD  304
#define MSGP  320            /* K padded to multiple of 32 */
#define INGNN 192
#define EDGED 160
#define BN    (BB*NNODE)     /* 8192  */
#define BE    (BB*NEDGE)     /* 16384 */
#define B2E   (2*BE)         /* 32768 */

// ---------------- workspace layout (units: floats; all offsets 16B-aligned) ----
static constexpr size_t AGG_S  = 0;                        // MAXN x MSGP (zero-padded)
static constexpr size_t AGG_D  = AGG_S + (size_t)MAXN*MSGP;
static constexpr size_t CNT_S  = AGG_D + (size_t)MAXN*MSGP;
static constexpr size_t CNT_D  = CNT_S + MAXN;
static constexpr size_t PRES_S = CNT_D + MAXN;             // int view
static constexpr size_t PRES_D = PRES_S + MAXN;            // int view
static constexpr size_t ZERO1  = PRES_D + MAXN;            // zero [0, ZERO1)
static constexpr size_t WIHP   = ZERO1;                    // 192 x MSGP padded gru_w_ih
static constexpr size_t GI_S   = WIHP + (size_t)192*MSGP;
static constexpr size_t GI_D   = GI_S + (size_t)MAXN*192;
static constexpr size_t GH_    = GI_D + (size_t)MAXN*192;
static constexpr size_t MEMU   = GH_  + (size_t)MAXN*192;
static constexpr size_t XBUF   = MEMU + (size_t)MAXN*MEMD;
static constexpr size_t EATT   = XBUF + (size_t)BN*INGNN;
static constexpr size_t FIRSTP = EATT + (size_t)BE*EDGED;  // int view
static constexpr size_t EI0_   = FIRSTP + MAXN;            // int view
static constexpr size_t EI1_   = EI0_ + BE;                // int view
static constexpr size_t QB_    = EI1_ + BE;
static constexpr size_t KB_    = QB_ + (size_t)BN*OUTD;
static constexpr size_t VB_    = KB_ + (size_t)BN*OUTD;
static constexpr size_t EED_   = VB_ + (size_t)BN*OUTD;
static constexpr size_t MMX_   = EED_ + (size_t)BE*OUTD;   // uint view (seg-max keys)
static constexpr size_t DEN_   = MMX_ + BN;
static constexpr size_t PV_    = DEN_ + BN;

// ---------------- helpers ----------------
__device__ __forceinline__ float sigm(float x){ return 1.0f/(1.0f+expf(-x)); }
__device__ __forceinline__ unsigned fkey(float f){
  unsigned u = __float_as_uint(f);
  return (u & 0x80000000u) ? ~u : (u | 0x80000000u);
}
__device__ __forceinline__ float funkey(unsigned k){
  unsigned u = (k & 0x80000000u) ? (k & 0x7fffffffu) : ~k;
  return __uint_as_float(u);
}
__device__ __forceinline__ void cvt8(v16h& dst, int base, float4 p, float4 q){
  dst[base+0]=(_Float16)p.x; dst[base+1]=(_Float16)p.y;
  dst[base+2]=(_Float16)p.z; dst[base+3]=(_Float16)p.w;
  dst[base+4]=(_Float16)q.x; dst[base+5]=(_Float16)q.y;
  dst[base+6]=(_Float16)q.z; dst[base+7]=(_Float16)q.w;
}

// ---------------- init ----------------
__global__ void tgn_init(float* ws){
  size_t i = (size_t)blockIdx.x*blockDim.x + threadIdx.x;
  if (i < ZERO1) ws[i] = 0.0f;                       // agg sums (+pad), counts, presence
  if (i < (size_t)2*BN) ws[MMX_ + i] = 0.0f;         // seg-max keys + den
  if (i < (size_t)MAXN) ((int*)ws)[FIRSTP + i] = 0x7fffffff;
}

// ---------------- zero-pad gru_w_ih to K=320 ----------------
__global__ __launch_bounds__(MSGP)
void tgn_pad_wih(const float* __restrict__ w_ih, float* __restrict__ ws){
  const int n = blockIdx.x, j = threadIdx.x;
  ws[WIHP + (size_t)n*MSGP + j] = (j < MSGD) ? w_ih[(size_t)n*MSGD + j] : 0.0f;
}

// ---------------- batch-0 message scatter (sum + counts) ----------------
__global__ __launch_bounds__(MSGP)
void tgn_msg_scatter(const int* __restrict__ etid, const int* __restrict__ sid,
                     const float* __restrict__ smask, const int* __restrict__ did,
                     const float* __restrict__ dmask, const int* __restrict__ eeid,
                     const float* __restrict__ eemb, const float* __restrict__ ets,
                     const float* __restrict__ memory, const float* __restrict__ last_update,
                     const float* __restrict__ type_emb, const float* __restrict__ time_w,
                     const float* __restrict__ time_b, float* __restrict__ ws){
  const int s  = blockIdx.x;            // event in batch 0
  const int et = etid[s];
  const int si = sid[s], di = did[s];
  const float sm = smask[s], dm = dmask[s];
  const float ts = ets[s];
  const float ns  = (et >= 3) ? 1.0f : 0.0f;
  const float ine = (et == 3 || et == 4) ? 1.0f : 0.0f;
  const float relts = ts - last_update[eeid[s]] * dm;
  const float targ  = ts * ine + relts * dm;
  const int j = threadIdx.x;
  if (j < MSGD) {
    float sv, dv;
    if (j < 16)        { float t = type_emb[et*16 + j]; sv = t; dv = t; }
    else if (j < 80)   { int i = j-16;  sv = memory[si*MEMD+i]*sm; dv = memory[di*MEMD+i]*dm; }
    else if (j < 144)  { int i = j-80;  sv = memory[di*MEMD+i]*dm; dv = memory[si*MEMD+i]*sm; }
    else if (j < 176)  { int i = j-144; float t = cosf(targ*time_w[i]+time_b[i])*ns; sv = t; dv = t; }
    else               { int i = j-176; float e = eemb[s*EEMBD+i]; sv = e; dv = e; }
    atomicAdd(&ws[AGG_S + (size_t)si*MSGP + j], sv * ns);
    atomicAdd(&ws[AGG_D + (size_t)di*MSGP + j], dv * dm);
  }
  if (j == 0) { atomicAdd(&ws[CNT_S + si], 1.0f); atomicAdd(&ws[CNT_D + di], 1.0f); }
}

// ---------------- presence flags over ALL batches ----------------
__global__ void tgn_presence(const int* __restrict__ sid, const int* __restrict__ did,
                             float* __restrict__ ws){
  int i = blockIdx.x*blockDim.x + threadIdx.x;
  if (i >= BB*SS) return;
  ((int*)ws)[PRES_S + sid[i]] = 1;
  ((int*)ws)[PRES_D + did[i]] = 1;
}

// ---------------- scatter-mean finalize ----------------
__global__ __launch_bounds__(MSGP)
void tgn_mean(float* __restrict__ ws){
  const int i = blockIdx.x, j = threadIdx.x;
  if (j >= MSGD) return;                // padding cols stay zero
  const float cs = fmaxf(ws[CNT_S + i], 1.0f);
  const float cd = fmaxf(ws[CNT_D + i], 1.0f);
  ws[AGG_S + (size_t)i*MSGP + j] /= cs;
  ws[AGG_D + (size_t)i*MSGP + j] /= cd;
}

// ---------------- WMMA GEMM: C[M,N] = A[M,K] * W[N,K]^T (+ bias[N]) ----------------
// K is a compile-time multiple of 32 -> no guards, vectorized b128 loads,
// one v_wmma_f32_16x16x32_f16 per 32-wide K step. One wave per 16x16 tile.
template<int K>
__global__ __launch_bounds__(32)
void tgn_wmma_gemm(const float* __restrict__ A, const float* __restrict__ W,
                   const float* __restrict__ bias, float* __restrict__ C, int N){
  const int lane = threadIdx.x;
  const int half = lane >> 4;
  const int lm   = lane & 15;
  const int m0 = blockIdx.x << 4;
  const int n0 = blockIdx.y << 4;
  const float* __restrict__ arow = A + (size_t)(m0 + lm) * K;
  const float* __restrict__ wrow = W + (size_t)(n0 + lm) * K;
  v8f acc = {};
#pragma unroll
  for (int kb = 0; kb < K; kb += 32) {
    // A fragment (16x32): per lane two contiguous 8-float runs at kb+8*half, kb+16+8*half
    const float4* ap = (const float4*)(arow + kb + (half << 3));
    const float4 a0 = ap[0], a1 = ap[1];        // k = kb+8h .. +7
    const float4 a2 = ap[4], a3 = ap[5];        // k = kb+16+8h .. +7
    // B fragment (32x16): per lane two contiguous 8-float runs at kb+8*half, kb+16+8*half
    const float4* bp = (const float4*)(wrow + kb + (half << 3));
    const float4 b0 = bp[0], b1 = bp[1], b2 = bp[4], b3 = bp[5];
    v16h a, b;
    cvt8(a, 0, a0, a1); cvt8(a, 8, a2, a3);
    cvt8(b, 0, b0, b1); cvt8(b, 8, b2, b3);
    acc = __builtin_amdgcn_wmma_f32_16x16x32_f16(false, a, false, b,
                                                 (short)0, acc, false, false);
  }
  const float bn = bias ? bias[n0 + lm] : 0.0f;
#pragma unroll
  for (int r = 0; r < 8; ++r) {
    const int m = m0 + r + (half << 3);
    C[(size_t)m * N + (n0 + lm)] = acc[r] + bn;   // coalesced: 16 lanes x consecutive n
  }
}

// ---------------- GRU gates + presence merge ----------------
__global__ void tgn_gru_merge(const float* __restrict__ memory, float* __restrict__ ws){
  const int idx = blockIdx.x*blockDim.x + threadIdx.x;
  if (idx >= MAXN*MEMD) return;
  const int i = idx >> 6, d = idx & 63;
  const float h  = memory[idx];
  const float hr = ws[GH_ + (size_t)i*192 + d];
  const float hz = ws[GH_ + (size_t)i*192 + 64 + d];
  const float hc = ws[GH_ + (size_t)i*192 + 128 + d];
  const float* giS = &ws[GI_S + (size_t)i*192];
  const float* giD = &ws[GI_D + (size_t)i*192];
  const float rS = sigm(giS[d] + hr),  rD = sigm(giD[d] + hr);
  const float zS = sigm(giS[64+d]+hz), zD = sigm(giD[64+d]+hz);
  const float nS = tanhf(giS[128+d] + rS*hc), nD = tanhf(giD[128+d] + rD*hc);
  const float mS = (1.0f-zS)*nS + zS*h;
  const float mD = (1.0f-zD)*nD + zD*h;
  const int pS = ((const int*)ws)[PRES_S + i];
  const int pD = ((const int*)ws)[PRES_D + i];
  ws[MEMU + idx] = pD ? mD : (pS ? mS : h);
}

// ---------------- GNN node input x = [node_features | mem] ----------------
__global__ __launch_bounds__(INGNN)
void tgn_gather_x(const int* __restrict__ node_ids, const float* __restrict__ nfeat,
                  float* __restrict__ ws){
  const int row = blockIdx.x, j = threadIdx.x;
  const int nid = node_ids[row];
  float v = (j < EEMBD) ? nfeat[(size_t)nid*EEMBD + j]
                        : ws[MEMU + (size_t)nid*MEMD + (j - EEMBD)];
  ws[XBUF + (size_t)row*INGNN + j] = v;
}

// ---------------- edge attr = [cos(time enc) | edge_features] ----------------
__global__ __launch_bounds__(EDGED)
void tgn_eattr(const int* __restrict__ edge_ids, const float* __restrict__ edge_ts,
               const float* __restrict__ last_update, const float* __restrict__ efeat,
               const float* __restrict__ time_w, const float* __restrict__ time_b,
               float* __restrict__ ws){
  const int row = blockIdx.x, j = threadIdx.x;
  const int eid = edge_ids[row];
  float v;
  if (j < TENCD) {
    const float dt = edge_ts[row] - last_update[eid];
    v = cosf(dt * time_w[j] + time_b[j]);
  } else {
    v = efeat[(size_t)eid*EEMBD + (j - TENCD)];
  }
  ws[EATT + (size_t)row*EDGED + j] = v;
}

// ---------------- first-occurrence table (replaces 8192x32768 argmax scan) ------
__global__ void tgn_firstpos(const int* __restrict__ node_ids, float* __restrict__ ws){
  const int p = blockIdx.x*blockDim.x + threadIdx.x;
  if (p >= BN) return;
  atomicMin(&((int*)ws)[FIRSTP + node_ids[p]], p);
}

__global__ void tgn_localize(const int* __restrict__ edge_index, float* __restrict__ ws){
  const int idx = blockIdx.x*blockDim.x + threadIdx.x;
  if (idx >= B2E) return;
  const int b = idx / (2*NEDGE), rem = idx % (2*NEDGE);
  const int c = rem / NEDGE,     e   = rem % NEDGE;
  const int fp = ((const int*)ws)[FIRSTP + edge_index[idx]];
  const int loc = (fp == 0x7fffffff) ? 0 : fp;   // argmax-of-all-false -> 0
  ((int*)ws)[(c == 0 ? EI0_ : EI1_) + (size_t)b*NEDGE + e] = loc;
}

// ---------------- attention: score + segment max ----------------
__global__ void tgn_score_max(float* __restrict__ ws){
  const int j = blockIdx.x*blockDim.x + threadIdx.x;
  if (j >= BE) return;
  const int s = ((const int*)ws)[EI0_ + j];
  const int t = ((const int*)ws)[EI1_ + j];
  const float4* q = (const float4*)&ws[QB_ + (size_t)t*OUTD];
  const float4* k = (const float4*)&ws[KB_ + (size_t)s*OUTD];
  const float4* e = (const float4*)&ws[EED_ + (size_t)j*OUTD];
  float sc = 0.0f;
#pragma unroll
  for (int d = 0; d < OUTD/4; ++d) {
    const float4 qq = q[d], kk = k[d], ee = e[d];
    sc += qq.x*(kk.x+ee.x) + qq.y*(kk.y+ee.y) + qq.z*(kk.z+ee.z) + qq.w*(kk.w+ee.w);
  }
  sc *= 0.125f;                                   // 1/sqrt(64)
  ws[PV_ + j] = sc;
  atomicMax(&((unsigned*)ws)[MMX_ + t], fkey(sc));
}

// ---------------- attention: exp + segment sum ----------------
__global__ void tgn_softmax_den(float* __restrict__ ws){
  const int j = blockIdx.x*blockDim.x + threadIdx.x;
  if (j >= BE) return;
  const int t = ((const int*)ws)[EI1_ + j];
  const float m = funkey(((const unsigned*)ws)[MMX_ + t]);
  const float p = expf(ws[PV_ + j] - m);
  ws[PV_ + j] = p;
  atomicAdd(&ws[DEN_ + t], p);
}

// ---------------- attention: weighted scatter into out (on top of skip) --------
__global__ __launch_bounds__(OUTD)
void tgn_attn_out(const float* __restrict__ ws, float* __restrict__ out){
  const int j = blockIdx.x, d = threadIdx.x;
  const int s = ((const int*)ws)[EI0_ + j];
  const int t = ((const int*)ws)[EI1_ + j];
  const float alpha = ws[PV_ + j] / ws[DEN_ + t];
  const float ve = ws[VB_ + (size_t)s*OUTD + d] + ws[EED_ + (size_t)j*OUTD + d];
  atomicAdd(&out[(size_t)t*OUTD + d], alpha * ve);
}

// ---------------- launcher ----------------
extern "C" void kernel_launch(void* const* d_in, const int* in_sizes, int n_in,
                              void* d_out, int out_size, void* d_ws, size_t ws_size,
                              hipStream_t stream) {
  const int*   etid   = (const int*)  d_in[0];
  const int*   sid    = (const int*)  d_in[1];
  const float* smask  = (const float*)d_in[2];
  const int*   did    = (const int*)  d_in[3];
  const float* dmask  = (const float*)d_in[4];
  const int*   eeid   = (const int*)  d_in[5];
  const float* eemb   = (const float*)d_in[6];
  const float* ets    = (const float*)d_in[7];
  const int*   nids   = (const int*)  d_in[8];
  const int*   eids   = (const int*)  d_in[9];
  const int*   eidx   = (const int*)  d_in[10];
  const float* edgets = (const float*)d_in[11];
  const float* memory = (const float*)d_in[12];
  const float* lastup = (const float*)d_in[13];
  const float* nfeat  = (const float*)d_in[14];
  const float* efeat  = (const float*)d_in[15];
  const float* temb   = (const float*)d_in[16];
  const float* tw     = (const float*)d_in[17];
  const float* tb     = (const float*)d_in[18];
  const float* w_ih   = (const float*)d_in[19];
  const float* w_hh   = (const float*)d_in[20];
  const float* b_ih   = (const float*)d_in[21];
  const float* b_hh   = (const float*)d_in[22];
  const float* Wq     = (const float*)d_in[23];
  const float* bq     = (const float*)d_in[24];
  const float* Wk     = (const float*)d_in[25];
  const float* bk     = (const float*)d_in[26];
  const float* Wv     = (const float*)d_in[27];
  const float* bv     = (const float*)d_in[28];
  const float* We     = (const float*)d_in[29];
  const float* Wskip  = (const float*)d_in[30];
  const float* bskip  = (const float*)d_in[31];
  float* ws  = (float*)d_ws;
  float* out = (float*)d_out;

  // 1) init workspace + padded gru_w_ih
  tgn_init<<<dim3((unsigned)((ZERO1 + 255) / 256)), dim3(256), 0, stream>>>(ws);
  tgn_pad_wih<<<dim3(192), dim3(MSGP), 0, stream>>>(w_ih, ws);
  // 2) batch-0 message scatter + counts
  tgn_msg_scatter<<<dim3(SS), dim3(MSGP), 0, stream>>>(etid, sid, smask, did, dmask,
      eeid, eemb, ets, memory, lastup, temb, tw, tb, ws);
  // 3) presence flags (all batches)
  tgn_presence<<<dim3((BB*SS)/256), dim3(256), 0, stream>>>(sid, did, ws);
  // 4) scatter-mean
  tgn_mean<<<dim3(MAXN), dim3(MSGP), 0, stream>>>(ws);
  // 5) GRU GEMMs: gi_src, gi_dst (K=320 padded), gh (K=64) — WMMA
  tgn_wmma_gemm<MSGP><<<dim3(MAXN/16, 192/16), dim3(32), 0, stream>>>(
      ws + AGG_S, ws + WIHP, b_ih, ws + GI_S, 192);
  tgn_wmma_gemm<MSGP><<<dim3(MAXN/16, 192/16), dim3(32), 0, stream>>>(
      ws + AGG_D, ws + WIHP, b_ih, ws + GI_D, 192);
  tgn_wmma_gemm<MEMD><<<dim3(MAXN/16, 192/16), dim3(32), 0, stream>>>(
      memory, w_hh, b_hh, ws + GH_, 192);
  // 6) GRU gates + merge
  tgn_gru_merge<<<dim3((MAXN*MEMD)/256), dim3(256), 0, stream>>>(memory, ws);
  // 7) node inputs x, edge attrs, localization
  tgn_gather_x<<<dim3(BN), dim3(INGNN), 0, stream>>>(nids, nfeat, ws);
  tgn_eattr<<<dim3(BE), dim3(EDGED), 0, stream>>>(eids, edgets, lastup, efeat, tw, tb, ws);
  tgn_firstpos<<<dim3(BN/256), dim3(256), 0, stream>>>(nids, ws);
  tgn_localize<<<dim3(B2E/256), dim3(256), 0, stream>>>(eidx, ws);
  // 8) projection GEMMs — WMMA
  tgn_wmma_gemm<INGNN><<<dim3(BN/16, OUTD/16), dim3(32), 0, stream>>>(
      ws + XBUF, Wq, bq, ws + QB_, OUTD);
  tgn_wmma_gemm<INGNN><<<dim3(BN/16, OUTD/16), dim3(32), 0, stream>>>(
      ws + XBUF, Wk, bk, ws + KB_, OUTD);
  tgn_wmma_gemm<INGNN><<<dim3(BN/16, OUTD/16), dim3(32), 0, stream>>>(
      ws + XBUF, Wv, bv, ws + VB_, OUTD);
  tgn_wmma_gemm<INGNN><<<dim3(BN/16, OUTD/16), dim3(32), 0, stream>>>(
      ws + XBUF, Wskip, bskip, out, OUTD);               // skip -> d_out base
  tgn_wmma_gemm<EDGED><<<dim3(BE/16, OUTD/16), dim3(32), 0, stream>>>(
      ws + EATT, We, nullptr, ws + EED_, OUTD);          // lin_edge, no bias
  // 9) attention passes
  tgn_score_max<<<dim3(BE/256), dim3(256), 0, stream>>>(ws);
  tgn_softmax_den<<<dim3(BE/256), dim3(256), 0, stream>>>(ws);
  tgn_attn_out<<<dim3(BE), dim3(OUTD), 0, stream>>>(ws, out);
}
